// LeNet_61194694034037
// MI455X (gfx1250) — matrix operand order repaired and checked by reading, not compiled, not run on hardware
//
#include <hip/hip_runtime.h>
#include <hip/hip_bf16.h>
#include <hip/hip_fp16.h>

// ---------------------------------------------------------------------------
// GCN (3 layers, shared edge structure) + LeNet FC head for MI455X (gfx1250).
// GCN stages: fp32, batch-innermost layout, bandwidth-bound by design.
// FC1/FC2: f16 WMMA (v_wmma_f32_16x16x32_f16), transposed GEMM C[M,B]=Wt x Ft.
// ---------------------------------------------------------------------------

typedef __attribute__((ext_vector_type(16))) _Float16 v16h;
typedef __attribute__((ext_vector_type(8)))  float    v8f;

union ABFrag { uint4 u[2]; v16h v; };
union OutPack { uint4 u; _Float16 h[8]; };

#define GNODES 231
#define GEDGES 1848

// ---------------------------------------------------------------------------
// CSR + norm prep: one block. deg -> inv_sqrt -> norm, CSR sorted by dst.
// edge_index is int64 per the reference (jnp.int64): [2][E], row0=src, row1=dst.
// Self loops appended as edges E..E+N-1.
// ---------------------------------------------------------------------------
__global__ void prep_csr_kernel(const long long* __restrict__ ei,
                                int* __restrict__ rowptr,
                                int* __restrict__ colidx,
                                float* __restrict__ val)
{
    const int E = GEDGES, N = GNODES;
    __shared__ int   degs[256];
    __shared__ int   fill[256];
    __shared__ float invs[256];
    __shared__ int   rps[GNODES + 1];
    int t = threadIdx.x;

    for (int i = t; i < N; i += blockDim.x) { degs[i] = 0; fill[i] = 0; }
    __syncthreads();

    const int total = E + N;
    for (int e = t; e < total; e += blockDim.x) {
        int d = (e < E) ? (int)ei[E + e] : (e - E);
        atomicAdd(&degs[d], 1);
    }
    __syncthreads();

    for (int i = t; i < N; i += blockDim.x)
        invs[i] = rsqrtf(fmaxf((float)degs[i], 1.0f));
    __syncthreads();

    if (t == 0) {
        int acc = 0;
        for (int n = 0; n < N; ++n) { rps[n] = acc; acc += degs[n]; }
        rps[N] = acc;
        for (int n = 0; n <= N; ++n) rowptr[n] = rps[n];
    }
    __syncthreads();

    for (int e = t; e < total; e += blockDim.x) {
        int s, d;
        if (e < E) { s = (int)ei[e]; d = (int)ei[E + e]; }
        else       { s = e - E; d = e - E; }
        float nm = invs[s] * invs[d];
        int pos = atomicAdd(&fill[d], 1);
        int idx = rps[d] + pos;
        colidx[idx] = s;
        val[idx]    = nm;
    }
}

// ---------------------------------------------------------------------------
// Tiled transpose: x[B][C] -> xt[C][B]   (C = N*12 = 2772)
// ---------------------------------------------------------------------------
__global__ void transpose_kernel(const float* __restrict__ in,
                                 float* __restrict__ out,
                                 int rows /*B*/, int cols /*C*/)
{
    __shared__ float tile[32][33];
    int c0 = blockIdx.x * 32, r0 = blockIdx.y * 32;
    int c = c0 + threadIdx.x;
    #pragma unroll
    for (int i = 0; i < 32; i += 8) {
        int r = r0 + threadIdx.y + i;
        if (r < rows && c < cols)
            tile[threadIdx.y + i][threadIdx.x] = in[(size_t)r * cols + c];
    }
    __syncthreads();
    int ob = r0 + threadIdx.x;           // batch index (coalesced on store)
    #pragma unroll
    for (int i = 0; i < 32; i += 8) {
        int oc = c0 + threadIdx.y + i;   // channel index
        if (oc < cols && ob < rows)
            out[(size_t)oc * rows + ob] = tile[threadIdx.x][threadIdx.y + i];
    }
}

// ---------------------------------------------------------------------------
// GCN layer, batch-innermost: in  = [N][FIN][B]  f32
//   agg[f] = sum_{e in CSR row n} val[e] * in[col[e]][f][b]
//   out[g] = act(bias[g] + sum_f agg[f]*W[f][g])
// ACT: 1 = tanh, store f32 [N][FOUT][B]
//      2 = none, store f16 into feat[b][480] at k = n*FOUT+g (FC input layout)
// ---------------------------------------------------------------------------
template<int FIN, int FOUT, int ACT>
__global__ void gcn_kernel(const float* __restrict__ hin,
                           void* __restrict__ hout,
                           const int* __restrict__ rowptr,
                           const int* __restrict__ colidx,
                           const float* __restrict__ val,
                           const float* __restrict__ W,
                           const float* __restrict__ bias,
                           int Bn)
{
    __shared__ float Ws[FIN * FOUT];
    __shared__ float bs[FOUT];
    for (int i = threadIdx.x; i < FIN * FOUT; i += blockDim.x) Ws[i] = W[i];
    for (int i = threadIdx.x; i < FOUT; i += blockDim.x) bs[i] = bias[i];
    __syncthreads();

    int b = blockIdx.x * blockDim.x + threadIdx.x;
    int n = blockIdx.y;
    if (b >= Bn) return;

    float agg[FIN];
    #pragma unroll
    for (int f = 0; f < FIN; ++f) agg[f] = 0.0f;

    int e0 = rowptr[n], e1 = rowptr[n + 1];
    for (int e = e0; e < e1; ++e) {
        int s = colidx[e];
        float w = val[e];
        const float* src = hin + (size_t)s * FIN * Bn + b;
        #pragma unroll
        for (int f = 0; f < FIN; ++f)
            agg[f] += w * src[(size_t)f * Bn];
    }

    #pragma unroll
    for (int g = 0; g < FOUT; ++g) {
        float o = bs[g];
        #pragma unroll
        for (int f = 0; f < FIN; ++f) o += agg[f] * Ws[f * FOUT + g];
        if (ACT == 1) {
            o = tanhf(o);
            ((float*)hout)[((size_t)n * FOUT + g) * Bn + b] = o;
        } else {
            ((_Float16*)hout)[(size_t)b * 480 + n * FOUT + g] = (_Float16)o;
        }
    }
}

// ---------------------------------------------------------------------------
// Zero-fill f16 buffer (feat K-padding rows 462..479 must be zero).
// ---------------------------------------------------------------------------
__global__ void zero_f16_kernel(_Float16* __restrict__ p, size_t n)
{
    size_t i = (size_t)blockIdx.x * blockDim.x + threadIdx.x;
    size_t stride = (size_t)gridDim.x * blockDim.x;
    for (; i < n; i += stride) p[i] = (_Float16)0.0f;
}

// ---------------------------------------------------------------------------
// WMMA GEMM:  Cout[col][m] = f16( ELU( bias[m] + sum_k Wt[m][k]*Bm[col][k] ) )
//   A (weights) = Wt[16 rows][KP] staged in LDS, from W[KREAL][MREAL] (row-major)
//   B (activations) = Bm[B rows][KP] f16, batch-major so each lane's 16-element
//     fragment is two contiguous 16-byte chunks (two b128 loads).
//   Grid: x = B/128 (8 waves x 16 cols each), y = MP/16.
//   EXEC is all-ones throughout (no divergence) as WMMA requires.
// Fragment maps (ISA 7.12.2, wave32, f16 16x16x32):
//   A: lane holds row=lane&15; a[i] = A[row][kb + klo + i (+8 if i>=8)], klo=8*(lane>=16)
//   B: lane holds col=lane&15; b[i] = B[kb + klo + i (+8 if i>=8)][col], klo=8*(lane>=16)
//   C: c[j] = C[mbase + 8*(lane>=16) + j][col]
// ---------------------------------------------------------------------------
template<int KP, int MP, int MREAL, int KREAL>
__global__ void gemm_wmma_kernel(const float* __restrict__ W,     // [KREAL][MREAL]
                                 const float* __restrict__ bias,  // [MREAL]
                                 const _Float16* __restrict__ Bm, // [Bn][KP]
                                 _Float16* __restrict__ Cout,     // [Bn][MP]
                                 int Bn)
{
    __shared__ __align__(32) _Float16 Wl[16 * KP];

    const int mbase = blockIdx.y * 16;

    // Stage transposed, padded weight slab: Wl[m][k] = W[k][mbase+m]
    for (int i = threadIdx.x; i < 16 * KP; i += blockDim.x) {
        int m = i / KP, k = i % KP;
        float v = 0.0f;
        if (k < KREAL && (mbase + m) < MREAL)
            v = W[(size_t)k * MREAL + (mbase + m)];
        Wl[i] = (_Float16)v;
    }
    __syncthreads();

    const int lane  = threadIdx.x & 31;
    const int wave  = threadIdx.x >> 5;
    const int ln15  = lane & 15;
    const int hi    = lane >> 4;
    const int klo   = hi ? 8 : 0;     // A/B-fragment K offset
    const int col   = blockIdx.x * 128 + wave * 16 + ln15;

    const _Float16* brow = Bm + (size_t)col * KP;

    v8f c = {};
    #pragma unroll
    for (int kb = 0; kb < KP; kb += 32) {
        ABFrag a, b;
        a.u[0] = *(const uint4*)(&Wl[ln15 * KP + kb + klo]);
        a.u[1] = *(const uint4*)(&Wl[ln15 * KP + kb + klo + 16]);
        b.u[0] = *(const uint4*)(brow + kb + klo);
        b.u[1] = *(const uint4*)(brow + kb + klo + 16);
        c = __builtin_amdgcn_wmma_f32_16x16x32_f16(
                /*neg_a=*/false, a.v, /*neg_b=*/false, b.v,
                /*c_mod=*/(short)0, c, /*reuse_a=*/false, /*reuse_b=*/false);
    }

    OutPack o;
    #pragma unroll
    for (int j = 0; j < 8; ++j) {
        int m = mbase + 8 * hi + j;
        float v = c[j] + ((m < MREAL) ? bias[m] : 0.0f);
        v = (v > 0.0f) ? v : (expf(v) - 1.0f);   // ELU; pad rows -> elu(0)=0
        o.h[j] = (_Float16)v;
    }
    *(uint4*)(Cout + (size_t)col * MP + mbase + 8 * hi) = o.u;
}

// ---------------------------------------------------------------------------
// FC3: out[b][j] = bf3[j] + sum_k act2[b][k] * Wf3[k][j]   (84 -> 10, fp32 out)
// ---------------------------------------------------------------------------
__global__ void fc3_kernel(const _Float16* __restrict__ act2, // [Bn][96]
                           const float* __restrict__ W,       // [84][10]
                           const float* __restrict__ bias,    // [10]
                           float* __restrict__ out,           // [Bn][10]
                           int Bn)
{
    __shared__ float Ws[84 * 10];
    __shared__ float bs[10];
    for (int i = threadIdx.x; i < 840; i += blockDim.x) Ws[i] = W[i];
    if (threadIdx.x < 10) bs[threadIdx.x] = bias[threadIdx.x];
    __syncthreads();

    int b = blockIdx.x * blockDim.x + threadIdx.x;
    if (b >= Bn) return;

    float acc[10];
    #pragma unroll
    for (int j = 0; j < 10; ++j) acc[j] = bs[j];

    const _Float16* a = act2 + (size_t)b * 96;
    for (int k = 0; k < 84; ++k) {
        float av = (float)a[k];
        #pragma unroll
        for (int j = 0; j < 10; ++j) acc[j] += av * Ws[k * 10 + j];
    }
    #pragma unroll
    for (int j = 0; j < 10; ++j) out[(size_t)b * 10 + j] = acc[j];
}

// ---------------------------------------------------------------------------
// Host launcher
// ---------------------------------------------------------------------------
extern "C" void kernel_launch(void* const* d_in, const int* in_sizes, int n_in,
                              void* d_out, int out_size, void* d_ws, size_t ws_size,
                              hipStream_t stream)
{
    const float*     x   = (const float*)d_in[0];
    const long long* ei  = (const long long*)d_in[1];   // int64 per reference
    const float* W1  = (const float*)d_in[2];
    const float* b1  = (const float*)d_in[3];
    const float* W2  = (const float*)d_in[4];
    const float* b2  = (const float*)d_in[5];
    const float* W3  = (const float*)d_in[6];
    const float* b3  = (const float*)d_in[7];
    const float* Wf1 = (const float*)d_in[8];
    const float* bf1 = (const float*)d_in[9];
    const float* Wf2 = (const float*)d_in[10];
    const float* bf2 = (const float*)d_in[11];
    const float* Wf3 = (const float*)d_in[12];
    const float* bf3 = (const float*)d_in[13];
    float* out = (float*)d_out;

    const int N = GNODES, E = GEDGES;
    const int Bn = in_sizes[0] / (N * 12);   // 16384 (multiple of 128)

    // Workspace carve (256B aligned partitions)
    char* p = (char*)d_ws;
    auto carve = [&](size_t bytes) -> char* {
        char* r = p;
        p += (bytes + 255) & ~(size_t)255;
        return r;
    };
    float*     xt   = (float*)carve((size_t)N * 12 * Bn * sizeof(float)); // x^T
    float*     h1t  = (float*)carve((size_t)N * 8  * Bn * sizeof(float));
    _Float16*  feat = (_Float16*)carve((size_t)Bn * 480 * sizeof(_Float16));
    _Float16*  act1 = (_Float16*)carve((size_t)Bn * 128 * sizeof(_Float16));
    _Float16*  act2 = (_Float16*)carve((size_t)Bn * 96  * sizeof(_Float16));
    int*       rowptr = (int*)carve((N + 1) * sizeof(int));
    int*       colidx = (int*)carve((E + N) * sizeof(int));
    float*     val    = (float*)carve((E + N) * sizeof(float));
    float*     h2t = xt;   // alias: xt is dead after GCN layer 1

    // 1) Graph structure prep (tiny)
    prep_csr_kernel<<<1, 256, 0, stream>>>(ei, rowptr, colidx, val);

    // 2) Transpose x [B][2772] -> xt [2772][B]
    {
        dim3 tb(32, 8);
        dim3 tg((N * 12 + 31) / 32, (Bn + 31) / 32);
        transpose_kernel<<<tg, tb, 0, stream>>>(x, xt, Bn, N * 12);
    }

    // 3) GCN layers (batch-innermost, coalesced gathers)
    gcn_kernel<12, 8, 1><<<dim3(Bn / 256, N), 256, 0, stream>>>(
        xt, h1t, rowptr, colidx, val, W1, b1, Bn);
    gcn_kernel<8, 4, 1><<<dim3(Bn / 256, N), 256, 0, stream>>>(
        h1t, h2t, rowptr, colidx, val, W2, b2, Bn);

    // 4) feat [B][480] f16 (zero K-padding), GCN3 fills k = n*2+g
    zero_f16_kernel<<<1024, 256, 0, stream>>>(feat, (size_t)Bn * 480);
    gcn_kernel<4, 2, 2><<<dim3(Bn / 256, N), 256, 0, stream>>>(
        h2t, (void*)feat, rowptr, colidx, val, W3, b3, Bn);

    // 5) FC1: [B,462]x[462,120] via WMMA (K pad 480, M pad 128), ELU -> act1
    gemm_wmma_kernel<480, 128, 120, 462><<<dim3(Bn / 128, 8), 256, 0, stream>>>(
        Wf1, bf1, feat, act1, Bn);

    // 6) FC2: [B,120]x[120,84] via WMMA (K pad 128, M pad 96), ELU -> act2
    gemm_wmma_kernel<128, 96, 84, 120><<<dim3(Bn / 128, 6), 256, 0, stream>>>(
        Wf2, bf2, act1, act2, Bn);

    // 7) FC3: [B,84]x[84,10] plain VALU (27 MFLOP), fp32 out
    fc3_kernel<<<Bn / 256, 256, 0, stream>>>(act2, Wf3, bf3, out, Bn);
}
